// MPGNN_PE_65893388256022
// MI455X (gfx1250) — compile-verified, matrix operand order checked by reading, not run on hardware
//
#include <hip/hip_runtime.h>
#include <hip/hip_bf16.h>

// ---------------------------------------------------------------------------
// MPGNN on gfx1250: f16 WMMA (v_wmma_f32_16x16x32_f16) with fp32 accumulate.
// ---------------------------------------------------------------------------

typedef __attribute__((ext_vector_type(16))) _Float16 v16h;
typedef __attribute__((ext_vector_type(8)))  _Float16 v8h;
typedef __attribute__((ext_vector_type(8)))  float    v8f;

static constexpr int NN = 50000;    // nodes
static constexpr int NE = 640000;   // edges
static constexpr int NG = 256;      // graphs
static constexpr int HD = 128;      // hidden
static constexpr int NL = 4;        // layers

__device__ __forceinline__ v8f wmma16(v16h a, v16h b, v8f c) {
  return __builtin_amdgcn_wmma_f32_16x16x32_f16(false, a, false, b,
                                                (short)0, c, false, false);
}

// 16 contiguous f32 (lo[0..7], hi[0..7]) -> v16h fragment
__device__ __forceinline__ v16h cvtf(const float* lo, const float* hi) {
  v16h a;
#pragma unroll
  for (int i = 0; i < 8; ++i) a[i] = (_Float16)lo[i];
#pragma unroll
  for (int i = 0; i < 8; ++i) a[i + 8] = (_Float16)hi[i];
  return a;
}
// 8 f32 in lo, upper half of K-chunk zero
__device__ __forceinline__ v16h cvtfz(const float* lo) {
  v16h a;
#pragma unroll
  for (int i = 0; i < 8; ++i) a[i] = (_Float16)lo[i];
#pragma unroll
  for (int i = 0; i < 8; ++i) a[i + 8] = (_Float16)0.f;
  return a;
}
// two contiguous v8h halves -> v16h fragment
__device__ __forceinline__ v16h ldfrag(const _Float16* rp) {
  v8h lo = *(const v8h*)rp;
  v8h hi = *(const v8h*)(rp + 16);
  return __builtin_shufflevector(lo, hi, 0, 1, 2, 3, 4, 5, 6, 7,
                                 8, 9, 10, 11, 12, 13, 14, 15);
}

// ---------------------------------------------------------------------------
// Pack fp32 weight [Ksrc x 128] into per-lane WMMA B-fragment layout:
// dst[((c*8 + ntile)*32 + lane)*16 + j], zero-padded for k >= Ksrc.
// Lane L holds column n = L%16; element j=2v+odd maps to
// k = 32c + (v<4 ? h*8 + 2v + odd : 16 + h*8 + 2(v-4) + odd), h = L/16.
// ---------------------------------------------------------------------------
__global__ void pack_weight_kernel(const float* __restrict__ src,
                                   _Float16* __restrict__ dst,
                                   int Ksrc, int chunks) {
  int idx = blockIdx.x * blockDim.x + threadIdx.x;
  int total = chunks * 4096;
  if (idx >= total) return;
  int j    = idx & 15;
  int lane = (idx >> 4) & 31;
  int t    = (idx >> 9) & 7;
  int c    = idx >> 12;
  int n = lane & 15, hh = lane >> 4;
  int v = j >> 1, odd = j & 1;
  int k = c * 32 + (v < 4 ? hh * 8 + 2 * v + odd
                          : 16 + hh * 8 + 2 * (v - 4) + odd);
  int col = t * 16 + n;
  float val = (k < Ksrc) ? src[k * HD + col] : 0.f;
  dst[idx] = (_Float16)val;
}

// ---------------------------------------------------------------------------
// Node/edge embedding GEMMs
// ---------------------------------------------------------------------------
__global__ __launch_bounds__(256) void embed_h_kernel(
    const float* __restrict__ x, const float* __restrict__ p,
    const _Float16* __restrict__ WP, const float* __restrict__ bias,
    _Float16* __restrict__ hout, int n_nodes) {
  const int tid = threadIdx.x, lane = tid & 31, wave = tid >> 5;
  const int nt = wave, n = lane & 15, hh = lane >> 4;
  v16h bf[3];
#pragma unroll
  for (int c = 0; c < 3; ++c)
    bf[c] = *(const v16h*)(WP + ((c * 8 + nt) * 32 + lane) * 16);
  const float bcol = bias[nt * 16 + n];
  const int ntiles = n_nodes / 16;
  for (int t = blockIdx.x; t < ntiles; t += gridDim.x) {
    const int row = t * 16 + n;  // A row m = lane%16
    v8f acc = {};
#pragma unroll
    for (int c = 0; c < 2; ++c) {
      const float* rp = x + row * 64 + c * 32 + hh * 8;
      acc = wmma16(cvtf(rp, rp + 16), bf[c], acc);
    }
    acc = wmma16(cvtfz(p + row * 16 + hh * 8), bf[2], acc);  // K=64..95 (pad)
    const int col = nt * 16 + n;
#pragma unroll
    for (int r = 0; r < 8; ++r)
      hout[(t * 16 + r + 8 * hh) * HD + col] = (_Float16)(acc[r] + bcol);
  }
}

__global__ __launch_bounds__(256) void embed_e_kernel(
    const float* __restrict__ e_attr, const _Float16* __restrict__ WP,
    const float* __restrict__ bias, _Float16* __restrict__ ef, int n_edges) {
  const int tid = threadIdx.x, lane = tid & 31, wave = tid >> 5;
  const int nt = wave, n = lane & 15, hh = lane >> 4;
  v16h bf = *(const v16h*)(WP + (nt * 32 + lane) * 16);
  const float bcol = bias[nt * 16 + n];
  const int ntiles = n_edges / 16;
  for (int t = blockIdx.x; t < ntiles; t += gridDim.x) {
    const int row = t * 16 + n;
    v8f acc = {};
    acc = wmma16(cvtfz(e_attr + row * 16 + hh * 8), bf, acc);  // K=0..31 (pad)
    const int col = nt * 16 + n;
#pragma unroll
    for (int r = 0; r < 8; ++r)
      ef[(t * 16 + r + 8 * hh) * HD + col] = (_Float16)(acc[r] + bcol);
  }
}

// ---------------------------------------------------------------------------
// Edge kernel: A = cat(h[send], h[rec], e) [16 x 384] staged in LDS;
// 16 waves = {Wm -> messages(scatter-add), We -> e_new} x 8 N-tiles.
// B fragments (12 chunks) held in registers across the grid-stride loop.
// ---------------------------------------------------------------------------
__global__ __launch_bounds__(512) void edge_kernel(
    const _Float16* __restrict__ hf, _Float16* ef,
    const int* __restrict__ edge_index,
    const _Float16* __restrict__ WmP, const _Float16* __restrict__ WeP,
    const float* __restrict__ bm, const float* __restrict__ be,
    float* __restrict__ agg, int n_edges) {
  constexpr int KCH  = 12;   // 384/32
  constexpr int ROWH = 392;  // padded LDS row stride (halves)
  __shared__ __align__(16) _Float16 tile[16 * ROWH];

  const int tid = threadIdx.x, lane = tid & 31, wave = tid >> 5;
  const int nt = wave & 7, isE = wave >> 3;
  const int n = lane & 15, hh = lane >> 4;

  const _Float16* Bp = isE ? WeP : WmP;
  v16h bf[KCH];
#pragma unroll
  for (int c = 0; c < KCH; ++c)
    bf[c] = *(const v16h*)(Bp + ((c * 8 + nt) * 32 + lane) * 16);
  const float bcol = (isE ? be : bm)[nt * 16 + n];

  const int ntiles = n_edges / 16;
  for (int t = blockIdx.x; t < ntiles; t += gridDim.x) {
    const int ebase = t * 16;
    // cooperative gather: 16 rows x 48 x (16B) loads
    for (int f = tid; f < 16 * 48; f += 512) {
      const int row = f / 48, q = f - row * 48;
      const _Float16* src;
      if (q < 16) {
        const int s = edge_index[ebase + row];            // send
        src = hf + s * HD + q * 8;
      } else if (q < 32) {
        const int r = edge_index[n_edges + ebase + row];  // rec
        src = hf + r * HD + (q - 16) * 8;
      } else {
        src = ef + (ebase + row) * HD + (q - 32) * 8;     // e
      }
      *(v8h*)(&tile[row * ROWH + q * 8]) = *(const v8h*)src;
    }
    __syncthreads();

    v8f acc = {};
#pragma unroll
    for (int c = 0; c < KCH; ++c) {
      const _Float16* rp = &tile[n * ROWH + c * 32 + hh * 8];
      acc = wmma16(ldfrag(rp), bf[c], acc);
    }
    __syncthreads();  // tile reads done before writes / next-iter overwrite

    const int col = nt * 16 + n;
    if (isE == 0) {
#pragma unroll
      for (int r = 0; r < 8; ++r) {
        const int m = r + 8 * hh;
        const int rec = edge_index[n_edges + ebase + m];
        atomicAdd(&agg[rec * HD + col], acc[r] + bcol);
      }
    } else {
#pragma unroll
      for (int r = 0; r < 8; ++r) {
        const int m = r + 8 * hh;
        ef[(ebase + m) * HD + col] = (_Float16)(acc[r] + bcol);
      }
    }
  }
}

// ---------------------------------------------------------------------------
// Node kernel: h_new = cat(h, agg) @ Wh + bh.  A rows are contiguous nodes;
// h read as f16, agg converted f32->f16 in-register. Ping-pong h buffers.
// ---------------------------------------------------------------------------
__global__ __launch_bounds__(256) void node_kernel(
    const _Float16* __restrict__ hin, const float* __restrict__ agg,
    const _Float16* __restrict__ WhP, const float* __restrict__ bh,
    _Float16* __restrict__ hout, float* __restrict__ hfp, int n_nodes) {
  constexpr int KCH = 8;  // 256/32
  const int tid = threadIdx.x, lane = tid & 31, wave = tid >> 5;
  const int nt = wave, n = lane & 15, hh = lane >> 4;
  v16h bf[KCH];
#pragma unroll
  for (int c = 0; c < KCH; ++c)
    bf[c] = *(const v16h*)(WhP + ((c * 8 + nt) * 32 + lane) * 16);
  const float bcol = bh[nt * 16 + n];

  const int ntiles = n_nodes / 16;
  for (int t = blockIdx.x; t < ntiles; t += gridDim.x) {
    const int row = t * 16 + n;
    v8f acc = {};
#pragma unroll
    for (int c = 0; c < 4; ++c) {  // K 0..127 from h (f16)
      const _Float16* rp = hin + row * HD + c * 32 + hh * 8;
      acc = wmma16(ldfrag(rp), bf[c], acc);
    }
#pragma unroll
    for (int c = 0; c < 4; ++c) {  // K 128..255 from agg (f32)
      const float* rp = agg + row * HD + c * 32 + hh * 8;
      acc = wmma16(cvtf(rp, rp + 16), bf[4 + c], acc);
    }
    const int col = nt * 16 + n;
#pragma unroll
    for (int r = 0; r < 8; ++r) {
      const int m = r + 8 * hh;
      const float v = acc[r] + bcol;
      hfp[(t * 16 + m) * HD + col] = v;
      hout[(t * 16 + m) * HD + col] = (_Float16)v;
    }
  }
}

// ---------------------------------------------------------------------------
// Pool: out[batch[i]] += h[i]; batch is sorted -> run-length local sums,
// one atomic per (graph boundary, column).
// ---------------------------------------------------------------------------
__global__ void pool_kernel(const float* __restrict__ hfp,
                            const int* __restrict__ batch,
                            float* __restrict__ out, int n_nodes, int chunk) {
  const int col = threadIdx.x;  // 128
  const int s = blockIdx.x * chunk;
  if (s >= n_nodes) return;
  const int e = min(s + chunk, n_nodes);
  float sum = 0.f;
  int cur = batch[s];
  for (int i = s; i < e; ++i) {
    int b = batch[i];
    if (b != cur) {
      atomicAdd(&out[cur * HD + col], sum);
      sum = 0.f;
      cur = b;
    }
    sum += hfp[i * HD + col];
  }
  atomicAdd(&out[cur * HD + col], sum);
}

// ---------------------------------------------------------------------------
extern "C" void kernel_launch(void* const* d_in, const int* in_sizes, int n_in,
                              void* d_out, int out_size, void* d_ws,
                              size_t ws_size, hipStream_t stream) {
  const float* x        = (const float*)d_in[0];
  const float* p        = (const float*)d_in[1];
  const float* e_attr   = (const float*)d_in[2];
  const int*   eidx     = (const int*)d_in[3];
  const int*   batch    = (const int*)d_in[4];
  const float* W_embed  = (const float*)d_in[5];
  const float* b_embed  = (const float*)d_in[6];
  const float* W_eembed = (const float*)d_in[7];
  const float* b_eembed = (const float*)d_in[8];
  const float* Wm       = (const float*)d_in[9];
  const float* bm       = (const float*)d_in[10];
  const float* Wh       = (const float*)d_in[11];
  const float* bh       = (const float*)d_in[12];
  const float* We       = (const float*)d_in[13];
  const float* be       = (const float*)d_in[14];
  float* out = (float*)d_out;

  char* ws = (char*)d_ws;
  size_t off = 0;
  auto wsalloc = [&](size_t bytes) -> void* {
    void* pr = ws + off;
    off = (off + bytes + 255) & ~(size_t)255;
    return pr;
  };
  _Float16* hb0 = (_Float16*)wsalloc((size_t)NN * HD * 2);
  _Float16* hb1 = (_Float16*)wsalloc((size_t)NN * HD * 2);
  _Float16* ef  = (_Float16*)wsalloc((size_t)NE * HD * 2);
  float*    agg = (float*)wsalloc((size_t)NN * HD * 4);
  float*    hfp = (float*)wsalloc((size_t)NN * HD * 4);
  _Float16* WembP = (_Float16*)wsalloc((size_t)3 * 4096 * 2);
  _Float16* WeeP  = (_Float16*)wsalloc((size_t)1 * 4096 * 2);
  _Float16* WmP[NL], *WeP[NL], *WhP[NL];
  for (int l = 0; l < NL; ++l) {
    WmP[l] = (_Float16*)wsalloc((size_t)12 * 4096 * 2);
    WeP[l] = (_Float16*)wsalloc((size_t)12 * 4096 * 2);
    WhP[l] = (_Float16*)wsalloc((size_t)8 * 4096 * 2);
  }

  // ---- pack weights into WMMA B-fragment layout (f16) ----
  auto pack = [&](const float* src, _Float16* dst, int Ksrc, int chunks) {
    int total = chunks * 4096;
    pack_weight_kernel<<<(total + 255) / 256, 256, 0, stream>>>(src, dst, Ksrc,
                                                                chunks);
  };
  pack(W_embed, WembP, 80, 3);
  pack(W_eembed, WeeP, 16, 1);
  for (int l = 0; l < NL; ++l) {
    pack(Wm + (size_t)l * 384 * HD, WmP[l], 384, 12);
    pack(We + (size_t)l * 384 * HD, WeP[l], 384, 12);
    pack(Wh + (size_t)l * 256 * HD, WhP[l], 256, 8);
  }

  // ---- embeddings ----
  embed_h_kernel<<<512, 256, 0, stream>>>(x, p, WembP, b_embed, hb0, NN);
  embed_e_kernel<<<512, 256, 0, stream>>>(e_attr, WeeP, b_eembed, ef, NE);

  // ---- layers ----
  _Float16* hbuf[2] = {hb0, hb1};
  for (int l = 0; l < NL; ++l) {
    hipMemsetAsync(agg, 0, (size_t)NN * HD * 4, stream);
    edge_kernel<<<512, 512, 0, stream>>>(hbuf[l & 1], ef, eidx, WmP[l], WeP[l],
                                         bm + l * HD, be + l * HD, agg, NE);
    node_kernel<<<512, 256, 0, stream>>>(hbuf[l & 1], agg, WhP[l], bh + l * HD,
                                         hbuf[(l + 1) & 1], hfp, NN);
  }

  // ---- global add pool ----
  hipMemsetAsync(out, 0, (size_t)NG * HD * 4, stream);
  const int chunk = 256;
  pool_kernel<<<(NN + chunk - 1) / chunk, 128, 0, stream>>>(hfp, batch, out, NN,
                                                            chunk);
}
